// LSTMModel_73194832658576
// MI455X (gfx1250) — compile-verified, matrix operand order checked by reading, not run on hardware
//
#include <hip/hip_runtime.h>

// ---------------------------------------------------------------------------
// Fused 3-layer LSTM (B=4096,T=128,F=32,H=64/32/16) + MLP head + softmax.
// One workgroup = 16 batch rows, 128 threads (4 wave32s).
// Gate GEMMs: v_wmma_f32_16x16x32_f16. Wave w owns the i/f/g/o gate tiles of
// hidden units [16w,16w+16), so the LSTM pointwise + cell state stay entirely
// in registers; only h (f16) round-trips LDS (ping-pong buffers, 4 barriers/t).
// Layer-1 weight B-tiles are hoisted into VGPRs for the whole time loop.
// Activations use v_exp_f32 + v_rcp_f32 (no IEEE division expansion).
// ---------------------------------------------------------------------------

typedef _Float16 half4 __attribute__((ext_vector_type(4)));
typedef _Float16 half8 __attribute__((ext_vector_type(8)));
typedef _Float16 v16h  __attribute__((ext_vector_type(16)));
typedef float    v4f   __attribute__((ext_vector_type(4)));
typedef float    v8f   __attribute__((ext_vector_type(8)));

namespace {
constexpr int Bsz = 4096;
constexpr int T   = 128;
constexpr int F   = 32;
constexpr int H1  = 64;
constexpr int H2  = 32;
constexpr int H3  = 16;   // padded to 32 in LDS
constexpr int NC  = 3;

constexpr int TILE = 16;   // batch rows per workgroup
constexpr int NTHR = 128;  // 4 wave32s

// LDS: f16 weights + x tile + double-buffered h1/h2/h3 (all 16B aligned)
constexpr int SMEM_BYTES = 2 * (256*32 + 256*64 + 128*64 + 128*32 + 64*32 + 64*32 +
                                16*32 + 2*16*64 + 2*16*32 + 2*16*32);  // = 91136
} // namespace

// Pack a 16x32 f16 operand tile from row-major LDS per CDNA5 wave32 layout:
//   lane<16 : K = {kc+0..7,  kc+16..23} of row (lane&15)
//   lane>=16: K = {kc+8..15, kc+24..31} of row (lane&15)
__device__ __forceinline__ v16h ld_tile(const _Float16* buf, int row_base,
                                        int stride, int kc, int lane) {
    const int r  = lane & 15;
    const int kb = (lane >> 4) << 3;  // 0 or 8
    const _Float16* p = buf + (row_base + r) * stride + kc + kb;
    union { v16h v; half8 h[2]; } u;
    u.h[0] = *(const half8*)(p);        // ds_load_b128
    u.h[1] = *(const half8*)(p + 16);   // ds_load_b128
    return u.v;
}

__device__ __forceinline__ v8f wmma_f16(v16h a, v16h b, v8f c) {
    return __builtin_amdgcn_wmma_f32_16x16x32_f16(
        false, a, false, b, (short)0, c, false, false);
}

// Fast activations: v_exp_f32 + v_rcp_f32, NaN-free for large |x|.
__device__ __forceinline__ float sig_(float x) {
    return __builtin_amdgcn_rcpf(1.0f + __expf(-x));
}
__device__ __forceinline__ float tanh_(float x) {
    return 1.0f - 2.0f * __builtin_amdgcn_rcpf(__expf(2.0f * x) + 1.0f);
}

// Register-resident LSTM pointwise on four D-layout gate tiles; stores h (f16)
// into LDS. acc[r] <-> batch row (r+mb), column col; c is the cell state tile.
__device__ __forceinline__ void lstm_pw(v8f aI, v8f aF, v8f aG, v8f aO,
                                        float bI, float bF, float bG, float bO,
                                        v8f& c, _Float16* hdst, int stride,
                                        int col, int mb) {
#pragma unroll
    for (int r = 0; r < 8; ++r) {
        const float iv = sig_(aI[r] + bI);
        const float fv = sig_(aF[r] + bF);
        const float gv = tanh_(aG[r] + bG);
        const float ov = sig_(aO[r] + bO);
        const float cc = fv * c[r] + iv * gv;
        c[r] = cc;
        hdst[(r + mb) * stride + col] = (_Float16)(ov * tanh_(cc));
    }
}

// Convert global f32 [R,C] -> LDS f16 [R,Cpad] (zero pad), cooperative.
__device__ __forceinline__ void cvt_w(_Float16* dst, const float* src,
                                      int R, int C, int Cpad, int tid) {
    for (int i = tid; i < R * Cpad; i += NTHR) {
        const int r = i / Cpad, c = i - r * Cpad;
        dst[i] = (_Float16)((c < C) ? src[r * C + c] : 0.0f);
    }
}

__global__ __launch_bounds__(NTHR, 1)
void lstm3_fused_kernel(const float* __restrict__ x,
                        const float* __restrict__ Wih1, const float* __restrict__ Whh1,
                        const float* __restrict__ bih1, const float* __restrict__ bhh1,
                        const float* __restrict__ Wih2, const float* __restrict__ Whh2,
                        const float* __restrict__ bih2, const float* __restrict__ bhh2,
                        const float* __restrict__ Wih3, const float* __restrict__ Whh3,
                        const float* __restrict__ bih3, const float* __restrict__ bhh3,
                        const float* __restrict__ Wfc1, const float* __restrict__ bfc1,
                        const float* __restrict__ Wfc2, const float* __restrict__ bfc2,
                        float* __restrict__ out) {
    extern __shared__ char smem[];
    _Float16* sWih1 = (_Float16*)smem;          // 256x32
    _Float16* sWhh1 = sWih1 + 256 * 32;         // 256x64
    _Float16* sWih2 = sWhh1 + 256 * 64;         // 128x64
    _Float16* sWhh2 = sWih2 + 128 * 64;         // 128x32
    _Float16* sWih3 = sWhh2 + 128 * 32;         // 64x32
    _Float16* sWhh3 = sWih3 + 64 * 32;          // 64x32 (cols>=16 zero)
    _Float16* sXt   = sWhh3 + 64 * 32;          // 16x32
    _Float16* sH1a  = sXt   + 16 * 32;          // 16x64 ping
    _Float16* sH1b  = sH1a  + 16 * 64;          // 16x64 pong
    _Float16* sH2a  = sH1b  + 16 * 64;          // 16x32
    _Float16* sH2b  = sH2a  + 16 * 32;
    _Float16* sH3a  = sH2b  + 16 * 32;          // 16x32 (cols>=16 zero)
    _Float16* sH3b  = sH3a  + 16 * 32;

    const int tid  = threadIdx.x;
    const int lane = tid & 31;
    const int wave = __builtin_amdgcn_readfirstlane(tid >> 5);  // uniform
    const int n    = lane & 15;              // column within a 16-wide tile
    const int mb   = (lane >> 4) << 3;       // row half offset (0 or 8)
    const int r0   = blockIdx.x * TILE;

    // ---- one-time setup: f16 weights in LDS, zero h buffers ----
    cvt_w(sWih1, Wih1, 4 * H1, F,  F,  tid);
    cvt_w(sWhh1, Whh1, 4 * H1, H1, H1, tid);
    cvt_w(sWih2, Wih2, 4 * H2, H1, H1, tid);
    cvt_w(sWhh2, Whh2, 4 * H2, H2, H2, tid);
    cvt_w(sWih3, Wih3, 4 * H3, H2, H2, tid);
    cvt_w(sWhh3, Whh3, 4 * H3, H3, 32, tid);   // K padded 16 -> 32
    for (int i = tid; i < 16 * 64; i += NTHR) {
        sH1a[i] = (_Float16)0.0f; sH1b[i] = (_Float16)0.0f;
    }
    for (int i = tid; i < 16 * 32; i += NTHR) {
        sH2a[i] = (_Float16)0.0f; sH2b[i] = (_Float16)0.0f;
        sH3a[i] = (_Float16)0.0f; sH3b[i] = (_Float16)0.0f;
    }
    __syncthreads();

    // ---- hoist layer-1 weight B-tiles into VGPRs for the whole time loop.
    // Gate block k (i,f,g,o) of units [16*wave,16*wave+16) is tile nt=4k+wave.
    v16h w1x[4], w1h0[4], w1h1[4];
#pragma unroll
    for (int k = 0; k < 4; ++k) {
        const int nt = 4 * k + wave;
        w1x[k]  = ld_tile(sWih1, nt * 16, F,  0,  lane);
        w1h0[k] = ld_tile(sWhh1, nt * 16, H1, 0,  lane);
        w1h1[k] = ld_tile(sWhh1, nt * 16, H1, 32, lane);
    }

    // ---- per-lane fused biases (column = owned hidden unit) ----
    const int j1 = wave * 16 + n;
    const float bI1 = bih1[j1]          + bhh1[j1];
    const float bF1 = bih1[H1 + j1]     + bhh1[H1 + j1];
    const float bG1 = bih1[2 * H1 + j1] + bhh1[2 * H1 + j1];
    const float bO1 = bih1[3 * H1 + j1] + bhh1[3 * H1 + j1];
    const int j2 = (wave & 1) * 16 + n;  // valid for wave<2
    const float bI2 = bih2[j2]          + bhh2[j2];
    const float bF2 = bih2[H2 + j2]     + bhh2[H2 + j2];
    const float bG2 = bih2[2 * H2 + j2] + bhh2[2 * H2 + j2];
    const float bO2 = bih2[3 * H2 + j2] + bhh2[3 * H2 + j2];
    const int j3 = n;                    // valid for wave==0
    const float bI3 = bih3[j3]          + bhh3[j3];
    const float bF3 = bih3[H3 + j3]     + bhh3[H3 + j3];
    const float bG3 = bih3[2 * H3 + j3] + bhh3[2 * H3 + j3];
    const float bO3 = bih3[3 * H3 + j3] + bhh3[3 * H3 + j3];

    // ---- register-resident cell state (D-tile layout, rows r+mb) ----
    const v8f Z = {0.f, 0.f, 0.f, 0.f, 0.f, 0.f, 0.f, 0.f};
    v8f c1 = Z, c2 = Z, c3 = Z;

    // x staging: one float4 per thread covers the whole 16x32 tile.
    const int xm = tid >> 3;             // batch row within tile
    const int xf = (tid & 7) << 2;       // feature quad
    const size_t xbase0 = ((size_t)(r0 + xm) * T) * F + xf;

    // ---- time loop ----
    for (int t = 0; t < T; ++t) {
        const _Float16* h1r = (t & 1) ? sH1b : sH1a;
        _Float16*       h1w = (t & 1) ? sH1a : sH1b;
        const _Float16* h2r = (t & 1) ? sH2b : sH2a;
        _Float16*       h2w = (t & 1) ? sH2a : sH2b;
        const _Float16* h3r = (t & 1) ? sH3b : sH3a;
        _Float16*       h3w = (t & 1) ? sH3a : sH3b;

        // stage x[:, t, :] as f16 (b128 load, b64 LDS store); prefetch t+1
        {
            const float* xp = x + xbase0 + (size_t)t * F;
            const v4f v = *(const v4f*)xp;
            half4 hv;
            hv.x = (_Float16)v.x; hv.y = (_Float16)v.y;
            hv.z = (_Float16)v.z; hv.w = (_Float16)v.w;
            *(half4*)(sXt + xm * F + xf) = hv;
            if (t + 1 < T) __builtin_prefetch(xp + F, 0, 1);
        }
        __syncthreads();

        // ---- layer 1: gates [16x256] = Xt(K=32) + h1(K=64); all 4 waves ----
        {
            const v16h aX = ld_tile(sXt, 0, F,  0,  lane);
            const v16h a0 = ld_tile(h1r, 0, H1, 0,  lane);
            const v16h a1 = ld_tile(h1r, 0, H1, 32, lane);
            v8f acc[4];
#pragma unroll
            for (int k = 0; k < 4; ++k) {
                v8f a = Z;
                a = wmma_f16(aX, w1x[k],  a);
                a = wmma_f16(a0, w1h0[k], a);
                a = wmma_f16(a1, w1h1[k], a);
                acc[k] = a;
            }
            lstm_pw(acc[0], acc[1], acc[2], acc[3], bI1, bF1, bG1, bO1,
                    c1, h1w, H1, j1, mb);
        }
        __syncthreads();

        // ---- layer 2: gates [16x128] = h1_new(K=64) + h2(K=32); waves 0,1 ----
        if (wave < 2) {
            const v16h a0 = ld_tile(h1w, 0, H1, 0,  lane);
            const v16h a1 = ld_tile(h1w, 0, H1, 32, lane);
            const v16h a2 = ld_tile(h2r, 0, H2, 0,  lane);
            v8f acc[4];
#pragma unroll
            for (int k = 0; k < 4; ++k) {
                const int nt = 2 * k + wave;
                v8f a = Z;
                a = wmma_f16(a0, ld_tile(sWih2, nt * 16, H1, 0,  lane), a);
                a = wmma_f16(a1, ld_tile(sWih2, nt * 16, H1, 32, lane), a);
                a = wmma_f16(a2, ld_tile(sWhh2, nt * 16, H2, 0,  lane), a);
                acc[k] = a;
            }
            lstm_pw(acc[0], acc[1], acc[2], acc[3], bI2, bF2, bG2, bO2,
                    c2, h2w, H2, j2, mb);
        }
        __syncthreads();

        // ---- layer 3: gates [16x64] = h2_new(K=32) + h3(K=16 pad); wave 0 ----
        if (wave == 0) {
            const v16h a0 = ld_tile(h2w, 0, H2, 0, lane);
            const v16h a1 = ld_tile(h3r, 0, 32, 0, lane);
            v8f acc[4];
#pragma unroll
            for (int k = 0; k < 4; ++k) {
                v8f a = Z;
                a = wmma_f16(a0, ld_tile(sWih3, k * 16, H2, 0, lane), a);
                a = wmma_f16(a1, ld_tile(sWhh3, k * 16, 32, 0, lane), a);
                acc[k] = a;
            }
            lstm_pw(acc[0], acc[1], acc[2], acc[3], bI3, bF3, bG3, bO3,
                    c3, h3w, 32, j3, mb);   // cols>=16 of h3 stay zero
        }
        __syncthreads();
    }

    // ---- MLP head on final h3 (written at t=127 -> buffer 'a') ----
    if (tid < TILE) {
        const int m = tid;
        float h[H3];
#pragma unroll
        for (int k = 0; k < H3; ++k) h[k] = (float)sH3a[m * 32 + k];
        float a1[8];
#pragma unroll
        for (int j = 0; j < 8; ++j) {
            float s = bfc1[j];
#pragma unroll
            for (int k = 0; k < H3; ++k) s += Wfc1[j * H3 + k] * h[k];
            a1[j] = fmaxf(s, 0.0f);
        }
        float lg[NC];
#pragma unroll
        for (int c = 0; c < NC; ++c) {
            float s = bfc2[c];
#pragma unroll
            for (int k = 0; k < 8; ++k) s += Wfc2[c * 8 + k] * a1[k];
            lg[c] = s;
        }
        const float mx = fmaxf(lg[0], fmaxf(lg[1], lg[2]));
        float e[NC], sum = 0.0f;
#pragma unroll
        for (int c = 0; c < NC; ++c) { e[c] = __expf(lg[c] - mx); sum += e[c]; }
        const float inv = __builtin_amdgcn_rcpf(sum);
#pragma unroll
        for (int c = 0; c < NC; ++c) out[(size_t)(r0 + m) * NC + c] = e[c] * inv;
    }
}

extern "C" void kernel_launch(void* const* d_in, const int* in_sizes, int n_in,
                              void* d_out, int out_size, void* d_ws, size_t ws_size,
                              hipStream_t stream) {
    (void)in_sizes; (void)n_in; (void)out_size; (void)d_ws; (void)ws_size;
    const float* x    = (const float*)d_in[0];
    const float* Wih1 = (const float*)d_in[1];
    const float* Whh1 = (const float*)d_in[2];
    const float* bih1 = (const float*)d_in[3];
    const float* bhh1 = (const float*)d_in[4];
    const float* Wih2 = (const float*)d_in[5];
    const float* Whh2 = (const float*)d_in[6];
    const float* bih2 = (const float*)d_in[7];
    const float* bhh2 = (const float*)d_in[8];
    const float* Wih3 = (const float*)d_in[9];
    const float* Whh3 = (const float*)d_in[10];
    const float* bih3 = (const float*)d_in[11];
    const float* bhh3 = (const float*)d_in[12];
    const float* Wfc1 = (const float*)d_in[13];
    const float* bfc1 = (const float*)d_in[14];
    const float* Wfc2 = (const float*)d_in[15];
    const float* bfc2 = (const float*)d_in[16];
    float* out = (float*)d_out;

    hipFuncSetAttribute(reinterpret_cast<const void*>(lstm3_fused_kernel),
                        hipFuncAttributeMaxDynamicSharedMemorySize, SMEM_BYTES);

    dim3 grid(Bsz / TILE);   // 256 workgroups x 16 batch rows
    dim3 block(NTHR);        // 4 wave32s
    lstm3_fused_kernel<<<grid, block, SMEM_BYTES, stream>>>(
        x, Wih1, Whh1, bih1, bhh1, Wih2, Whh2, bih2, bhh2,
        Wih3, Whh3, bih3, bhh3, Wfc1, bfc1, Wfc2, bfc2, out);
}
